// Self_attention_48567490183565
// MI455X (gfx1250) — matrix operand order repaired and checked by reading, not compiled, not run on hardware
//
#include <hip/hip_runtime.h>

typedef __attribute__((ext_vector_type(16))) _Float16 v16h;
typedef __attribute__((ext_vector_type(8)))  _Float16 h8v;
typedef __attribute__((ext_vector_type(8)))  float    v8f;
typedef __attribute__((ext_vector_type(4)))  float    f4v;
typedef __attribute__((ext_vector_type(4)))  unsigned int u32x4;
typedef __attribute__((ext_vector_type(8)))  int      i32x8;
typedef __attribute__((ext_vector_type(4)))  int      i32x4;

#if __has_builtin(__builtin_amdgcn_tensor_load_to_lds) && \
    __has_builtin(__builtin_amdgcn_s_wait_tensorcnt)
#define HAVE_TDM 1
#else
#define HAVE_TDM 0
#endif

#define WMMA_F16(A, B, C) \
  __builtin_amdgcn_wmma_f32_16x16x32_f16(false, (A), false, (B), (short)0, (C), false, false)

__device__ __forceinline__ v16h hcomb(h8v lo, h8v hi) {
  v16h r;
#pragma unroll
  for (int i = 0; i < 8; ++i) { r[i] = lo[i]; r[i + 8] = hi[i]; }
  return r;
}

#if HAVE_TDM
// Issue one TDM 2D tile load: tile_d0 x tile_d1 elements (2-byte) from a
// row-major [tensor_d1][stride0] tensor into LDS (tile rows contiguous).
// D# bit layout per CDNA5 ISA ch.8 (groups 0/1; groups 2/3 zero for 2D).
__device__ __forceinline__ void tdm_load_2d(unsigned lds_addr, unsigned long long gaddr,
                                            unsigned tensor_d0, unsigned tensor_d1,
                                            unsigned tile_d0, unsigned tile_d1,
                                            unsigned long long stride0) {
  u32x4 g0;
  g0[0] = 1u;                                            // count=1, user descriptor
  g0[1] = lds_addr;                                      // LDS byte address
  g0[2] = (unsigned)(gaddr & 0xFFFFFFFFu);               // global_addr[31:0]
  g0[3] = (unsigned)((gaddr >> 32) & 0x01FFFFFFu)        // global_addr[56:32]
        | 0x80000000u;                                   // type=2 ("image")
  i32x8 g1;
  g1[0] = (int)(1u << 16);                               // data_size=1 (2 bytes)
  g1[1] = (int)((tensor_d0 & 0xFFFFu) << 16);            // tensor_dim0[15:0]
  g1[2] = (int)((tensor_d0 >> 16) | ((tensor_d1 & 0xFFFFu) << 16));
  g1[3] = (int)((tensor_d1 >> 16) | (tile_d0 << 16));    // tile_dim0
  g1[4] = (int)(tile_d1 & 0xFFFFu);                      // tile_dim1 (tile_dim2=0)
  g1[5] = (int)(unsigned)(stride0 & 0xFFFFFFFFu);        // dim0 stride[31:0]
  g1[6] = (int)(unsigned)((stride0 >> 32) & 0xFFFFu);    // dim0 stride[47:32]
  g1[7] = 0;
  const i32x4 gz4 = {0, 0, 0, 0};
  const i32x8 gz8 = {0, 0, 0, 0, 0, 0, 0, 0};
  __builtin_amdgcn_tensor_load_to_lds(g0, g1, gz4, gz4, gz8, 0);
}
#endif

// -------------------------------------------------------------------------
// Projection: rows [0,32) of W are q_w, rows [32,288) are v_w.
// out tiles 0..1 -> qh[b][n][32] (f16, transposed), tiles 2..17 -> vh[b][c][n]
// grid (n/16, 18, B), block 32 (one wave)
// -------------------------------------------------------------------------
__global__ __launch_bounds__(32) void proj_kernel(
    const float* __restrict__ x,
    const float* __restrict__ qw, const float* __restrict__ qb,
    const float* __restrict__ vw, const float* __restrict__ vbias,
    _Float16* __restrict__ qh, _Float16* __restrict__ vh)
{
  const int lane = threadIdx.x & 31;
  const int ln16 = lane & 15;
  const int hi16 = (lane >> 4) & 1;
  const int nt = blockIdx.x;
  const int ot = blockIdx.y;
  const int b  = blockIdx.z;
  const bool isq = (ot < 2);
  const int  out0 = isq ? ot * 16 : (ot - 2) * 16;
  const float* W  = isq ? qw : vw;
  const float* Bv = isq ? qb : vbias;
  const int n = nt * 16 + ln16;

  v8f acc = {0.f, 0.f, 0.f, 0.f, 0.f, 0.f, 0.f, 0.f};
#pragma unroll
  for (int kc = 0; kc < 8; ++kc) {
    const int f0 = kc * 32;
    const float* wrow = W + (size_t)(out0 + ln16) * 256 + f0 + hi16 * 8;
    f4v w0 = *(const f4v*)(wrow);
    f4v w1 = *(const f4v*)(wrow + 4);
    f4v w2 = *(const f4v*)(wrow + 16);
    f4v w3 = *(const f4v*)(wrow + 20);
    v16h a;
#pragma unroll
    for (int i = 0; i < 4; ++i) {
      a[i]      = (_Float16)w0[i];
      a[i + 4]  = (_Float16)w1[i];
      a[i + 8]  = (_Float16)w2[i];
      a[i + 12] = (_Float16)w3[i];
    }
    const float* xcol = x + ((size_t)b * 256 + f0 + hi16 * 8) * 4096 + n;
    v16h bt;
#pragma unroll
    for (int k = 0; k < 16; ++k) bt[k] = (_Float16)xcol[(size_t)(k + (k & 8)) * 4096];
    acc = WMMA_F16(a, bt, acc);
  }
#pragma unroll
  for (int r = 0; r < 8; ++r) {
    const int orow = out0 + r + hi16 * 8;
    const float val = acc[r] + Bv[orow];
    if (isq) qh[((size_t)b * 4096 + n) * 32 + orow] = (_Float16)val;
    else     vh[((size_t)b * 256 + orow) * 4096 + n] = (_Float16)val;
  }
}

// -------------------------------------------------------------------------
// Flash attention: grid (n/64, B), block 256 (8 waves).
// 64 queries x 256 channels per WG; K/V tiles double-buffered in LDS via the
// Tensor Data Mover (wave 0 issues, TENSORcnt-tracked, overlaps compute).
// Wave w: query tile (w>>1), channel half (w&1)*128.
// -------------------------------------------------------------------------
__global__ __launch_bounds__(256) void attn_kernel(
    const _Float16* __restrict__ qh, const _Float16* __restrict__ vh,
    const float* __restrict__ x, const float* __restrict__ gamma,
    float* __restrict__ out)
{
  __shared__ _Float16 klds[2][32 * 32];    // 32 keys x 32 feats      (2x2 KB)
  __shared__ _Float16 vlds[2][256 * 32];   // 256 ch  x 32 keys       (2x16 KB)
  __shared__ _Float16 pbuf[8][16 * 32];    // per-wave P staging      (8 KB)
  __shared__ float    obuf[8][16 * 16];    // per-wave O transpose    (8 KB)

  const int tid  = threadIdx.x;
  const int lane = tid & 31;
  const int wv   = tid >> 5;
  const int ln16 = lane & 15;
  const int hi16 = (lane >> 4) & 1;
  const int b    = blockIdx.y;
  const int qt   = blockIdx.x * 4 + (wv >> 1);   // this wave's query tile
  const int chw  = (wv & 1) * 128;               // this wave's channel base
  const float g  = gamma[0];

  const _Float16* kglob = qh + (size_t)b * 4096 * 32;   // [key][feat]
  const _Float16* vglob = vh + (size_t)b * 256 * 4096;  // [ch][key]

  // Q A-tile, resident all loop
  const _Float16* qbase = qh + ((size_t)b * 4096 + qt * 16 + ln16) * 32 + hi16 * 16;
  const v16h qa = hcomb(*(const h8v*)(qbase), *(const h8v*)(qbase + 8));

  v8f acc[8];
  const v8f vzero = {0.f, 0.f, 0.f, 0.f, 0.f, 0.f, 0.f, 0.f};
#pragma unroll
  for (int t = 0; t < 8; ++t) acc[t] = vzero;
  float m[8], l[8];
#pragma unroll
  for (int r = 0; r < 8; ++r) { m[r] = -1e30f; l[r] = 0.f; }

#if HAVE_TDM
  // Prologue: TDM-load tile 0 into buffer 0 (wave 0 issues; EXEC full).
  if (wv == 0) {
    tdm_load_2d((unsigned)(size_t)(void*)klds[0], (unsigned long long)(size_t)(kglob + 0 * 32),
                32u, 4096u, 32u, 32u, 32ull);
    tdm_load_2d((unsigned)(size_t)(void*)vlds[0], (unsigned long long)(size_t)(vglob + 0),
                4096u, 256u, 32u, 256u, 4096ull);
  }
#endif

  for (int jt = 0; jt < 128; ++jt) {
    const int j0  = jt * 32;
    const int cur = jt & 1;

#if HAVE_TDM
    __syncthreads();                       // all waves done with tile jt-1
    if (wv == 0) {
      if (jt + 1 < 128) {                  // prefetch tile jt+1 into other buffer
        const int nj = (jt + 1) * 32;
        tdm_load_2d((unsigned)(size_t)(void*)klds[cur ^ 1],
                    (unsigned long long)(size_t)(kglob + (size_t)nj * 32),
                    32u, 4096u, 32u, 32u, 32ull);
        tdm_load_2d((unsigned)(size_t)(void*)vlds[cur ^ 1],
                    (unsigned long long)(size_t)(vglob + nj),
                    4096u, 256u, 32u, 256u, 4096ull);
        __builtin_amdgcn_s_wait_tensorcnt(2);   // tile jt complete (in-order)
      } else {
        __builtin_amdgcn_s_wait_tensorcnt(0);
      }
    }
    __syncthreads();                       // tile jt visible to all waves
#else
    __syncthreads();
    if (tid < 128) {                       // K: 32x32, 128 b128 chunks
      const int key = tid >> 2, chunk = tid & 3;
      const h8v kv = *(const h8v*)(kglob + ((size_t)j0 + key) * 32 + chunk * 8);
      *(h8v*)(klds[cur] + key * 32 + chunk * 8) = kv;
    }
#pragma unroll
    for (int rep = 0; rep < 4; ++rep) {    // V: 256x32, 1024 b128 chunks
      const int li = rep * 256 + tid;
      const int ch = li >> 2, chunk = li & 3;
      const h8v vv = *(const h8v*)(vglob + (size_t)ch * 4096 + j0 + chunk * 8);
      *(h8v*)(vlds[cur] + ch * 32 + chunk * 8) = vv;
    }
    __syncthreads();
#endif

    // ---- S = Q K^T from LDS ----
    const _Float16* kr0 = klds[cur] + ln16 * 32 + hi16 * 16;
    const _Float16* kr1 = klds[cur] + (16 + ln16) * 32 + hi16 * 16;
    const v16h kb0 = hcomb(*(const h8v*)(kr0), *(const h8v*)(kr0 + 8));
    const v16h kb1 = hcomb(*(const h8v*)(kr1), *(const h8v*)(kr1 + 8));
    v8f s0 = WMMA_F16(qa, kb0, vzero);
    v8f s1 = WMMA_F16(qa, kb1, vzero);

    // ---- online softmax (rows r / r+8 per lane-half) ----
    float mt[8];
#pragma unroll
    for (int r = 0; r < 8; ++r) mt[r] = fmaxf(s0[r], s1[r]);
#pragma unroll
    for (int off = 8; off >= 1; off >>= 1)
#pragma unroll
      for (int r = 0; r < 8; ++r) mt[r] = fmaxf(mt[r], __shfl_xor(mt[r], off, 32));
    float sc[8], rs[8];
#pragma unroll
    for (int r = 0; r < 8; ++r) {
      const float mn = fmaxf(m[r], mt[r]);
      sc[r] = __expf(m[r] - mn);
      m[r] = mn;
      const float p0 = __expf(s0[r] - mn);
      const float p1 = __expf(s1[r] - mn);
      s0[r] = p0; s1[r] = p1;
      rs[r] = p0 + p1;
    }
#pragma unroll
    for (int off = 8; off >= 1; off >>= 1)
#pragma unroll
      for (int r = 0; r < 8; ++r) rs[r] += __shfl_xor(rs[r], off, 32);
#pragma unroll
    for (int r = 0; r < 8; ++r) l[r] = l[r] * sc[r] + rs[r];
#pragma unroll
    for (int t = 0; t < 8; ++t)
#pragma unroll
      for (int r = 0; r < 8; ++r) acc[t][r] *= sc[r];

    // ---- P (f32 C layout) -> LDS f16 row-major -> A layout (per-wave) ----
    _Float16* pb = pbuf[wv];
#pragma unroll
    for (int r = 0; r < 8; ++r) {
      const int row = r + hi16 * 8;
      pb[row * 32 + ln16]      = (_Float16)s0[r];
      pb[row * 32 + 16 + ln16] = (_Float16)s1[r];
    }
    asm volatile("s_wait_dscnt 0x0" ::: "memory");
    const _Float16* pr = pb + ln16 * 32 + hi16 * 16;
    const v16h pa = hcomb(*(const h8v*)(pr), *(const h8v*)(pr + 8));

    // ---- O += P * V^T : B from LDS ----
#pragma unroll
    for (int t = 0; t < 8; ++t) {
      const _Float16* vr = vlds[cur] + (chw + t * 16 + ln16) * 32 + hi16 * 16;
      const v16h vt = hcomb(*(const h8v*)(vr), *(const h8v*)(vr + 8));
      acc[t] = WMMA_F16(pa, vt, acc[t]);
    }
  }

  // normalize, transpose through LDS so stores are contiguous along i, residual
  float* ob = obuf[wv];
#pragma unroll
  for (int t = 0; t < 8; ++t) {
#pragma unroll
    for (int r = 0; r < 8; ++r)
      ob[ln16 * 16 + r + hi16 * 8] = acc[t][r] / l[r];   // [c_local][i_local]
    asm volatile("s_wait_dscnt 0x0" ::: "memory");
#pragma unroll
    for (int r = 0; r < 8; ++r) {
      const int cl = r + hi16 * 8;
      const size_t oi =
          ((size_t)b * 256 + chw + t * 16 + cl) * 4096 + (size_t)qt * 16 + ln16;
      out[oi] = g * ob[cl * 16 + ln16] + x[oi];
    }
    asm volatile("s_wait_dscnt 0x0" ::: "memory");       // before ob reuse
  }
}

extern "C" void kernel_launch(void* const* d_in, const int* in_sizes, int n_in,
                              void* d_out, int out_size, void* d_ws, size_t ws_size,
                              hipStream_t stream) {
  const float* x  = (const float*)d_in[0];
  const float* qw = (const float*)d_in[1];
  const float* qb = (const float*)d_in[2];
  const float* vw = (const float*)d_in[3];
  const float* vb = (const float*)d_in[4];
  const float* gm = (const float*)d_in[5];
  float* out = (float*)d_out;

  _Float16* qh = (_Float16*)d_ws;                                      // 8*4096*32 f16 = 2 MB
  _Float16* vh = (_Float16*)((char*)d_ws + (size_t)8 * 4096 * 32 * 2); // 8*256*4096 f16 = 16 MB

  proj_kernel<<<dim3(256, 18, 8), 32, 0, stream>>>(x, qw, qb, vw, vb, qh, vh);
  attn_kernel<<<dim3(64, 8), 256, 0, stream>>>(qh, vh, x, gm, out);
}
